// Infor_Inter_Layer_63376537419958
// MI455X (gfx1250) — matrix operand order repaired and checked by reading, not compiled, not run on hardware
//
#include <hip/hip_runtime.h>
#include <hip/hip_bf16.h>
#include <math.h>

// ---------------------------------------------------------------------------
// CDNA5 (gfx1250) fused GNN layer.  All GEMMs run on v_wmma_f32_16x16x32_bf16
// with weights resident in LDS (padded 120->128).  wave32 fragment layouts per
// CDNA5 ISA 7.12.2.  Weight fills use GLOBAL_LOAD_ASYNC_TO_LDS_B128 (ASYNCcnt).
// ---------------------------------------------------------------------------

typedef __attribute__((ext_vector_type(16))) __bf16 v16bf;
typedef __attribute__((ext_vector_type(8)))  float  v8f;
typedef __attribute__((__vector_size__(16))) int    v4i_t;

#define D_DIM   120
#define D_PAD   128
#define G_MAX   2048
#define R_DIM   15
#define MAT_ELEMS (D_PAD * D_PAD)   // 16384 bf16 per weight matrix
#define N_MATS  8                   // Wl, Whw, Wr, Wz, Wn, Whr, Whz, Whn

#if __has_builtin(__builtin_amdgcn_global_load_async_to_lds_b128)
#define HAVE_ASYNC_LDS 1
typedef __attribute__((address_space(1))) v4i_t* as1_v4i_p;
typedef __attribute__((address_space(3))) v4i_t* as3_v4i_p;
#endif

// Fast device math: v_rcp_f32 / v_exp_f32 based (no IEEE divide expansion).
__device__ __forceinline__ float fast_rcp(float x) { return __builtin_amdgcn_rcpf(x); }
__device__ __forceinline__ float sigf(float x) {
  return fast_rcp(1.0f + __expf(-x));
}
__device__ __forceinline__ float tanh_fast(float x) {
  // tanh(x) = 1 - 2/(exp(2x)+1); exp saturates cleanly for |x| large.
  return 1.0f - 2.0f * fast_rcp(__expf(2.0f * x) + 1.0f);
}

__device__ __forceinline__ v8f wmma_bf16(v16bf a, v16bf b, v8f c) {
  return __builtin_amdgcn_wmma_f32_16x16x32_bf16(false, a, false, b, (short)0, c,
                                                 false, false);
}

// Cooperative block fill of LDS from global (bytes % 4096 == 0, block=256).
__device__ __forceinline__ void lds_fill(void* dst_lds, const void* src_global,
                                         int bytes) {
  const int tid = threadIdx.x;
#if defined(HAVE_ASYNC_LDS)
  char* g = (char*)src_global;   // strip const for builtin arg conversion
  char* l = (char*)dst_lds;
  for (int off = tid * 16; off < bytes; off += 256 * 16) {
    __builtin_amdgcn_global_load_async_to_lds_b128(
        (as1_v4i_p)(g + off), (as3_v4i_p)(l + off), 0, 0);
  }
#if __has_builtin(__builtin_amdgcn_s_wait_asynccnt)
  __builtin_amdgcn_s_wait_asynccnt(0);
#else
  asm volatile("s_wait_asynccnt 0" ::: "memory");
#endif
#else
  const uint4* src = (const uint4*)src_global;
  uint4* dst = (uint4*)dst_lds;
  for (int i = tid; i < bytes / 16; i += 256) dst[i] = src[i];
#endif
}

// A-fragment (16x32 bf16) from a row-major f32 row pointer (per-lane row).
// ISA layout: lanes 0-15 -> M=lane, K={0..7,16..23}; lanes 16-31 -> K={8..15,24..31}.
__device__ __forceinline__ v16bf a_frag_f32(const float* rp, int kbase, int kmax) {
  const int half = (threadIdx.x >> 4) & 1;
  v16bf a;
#pragma unroll
  for (int v = 0; v < 8; ++v) {
    const int k = kbase + ((v & 4) ? 16 : 0) + half * 8 + (v & 3) * 2;
    const float x0 = (k     < kmax) ? rp[k]     : 0.0f;
    const float x1 = (k + 1 < kmax) ? rp[k + 1] : 0.0f;
    a[2 * v]     = (__bf16)x0;
    a[2 * v + 1] = (__bf16)x1;
  }
  return a;
}

// A-fragment from a wave-private LDS tile [16][128] bf16 (row-major).
// Per 32-wide K step each lane's data is two contiguous 16B runs.
__device__ __forceinline__ v16bf a_frag_lds(const __bf16* tile, int kbase) {
  const int lane = threadIdx.x & 31;
  const int half = lane >> 4;
  const __bf16* rp = tile + (lane & 15) * D_PAD + kbase + half * 8;
  v16bf a;
#pragma unroll
  for (int v = 0; v < 8; ++v) {
    a[v]     = rp[v];        // K = kbase + half*8 + v
    a[8 + v] = rp[16 + v];   // K = kbase + 16 + half*8 + v
  }
  return a;
}

// B-fragment (32x16 bf16) from an LDS weight matrix stored [n][k] (k contig).
// B[k][n]: lanes 0-15 -> K={0..7,16..23}, lanes 16-31 -> K={8..15,24..31}.
__device__ __forceinline__ v16bf b_frag(const __bf16* w, int kbase, int nt) {
  const int lane = threadIdx.x & 31;
  const int n  = nt * 16 + (lane & 15);
  const int kb = kbase + ((lane >> 4) << 3);
  const __bf16* rp = w + n * D_PAD + kb;   // two 16B runs -> 2x ds_load_b128
  v16bf b;
#pragma unroll
  for (int v = 0; v < 8; ++v) {
    b[v]     = rp[v];
    b[8 + v] = rp[16 + v];
  }
  return b;
}

// Write 8 C fragments (16 rows x 128 cols f32) into an LDS bf16 tile [16][128].
__device__ __forceinline__ void c_to_stage(__bf16* stage, const v8f* cf) {
  const int lane = threadIdx.x & 31;
  const int col  = lane & 15;
  const int half = lane >> 4;
#pragma unroll
  for (int nt = 0; nt < 8; ++nt) {
    const int ch = nt * 16 + col;
#pragma unroll
    for (int v = 0; v < 8; ++v) {
      stage[(v + 8 * half) * D_PAD + ch] = (__bf16)cf[nt][v];
    }
  }
}

// ---------------------------------------------------------------------------
// Kernel 0a: zero segment accumulators (sum_s | sumSig | g_feat | cnt).
// ---------------------------------------------------------------------------
__global__ void k_zero(float* p, int n) {
  int i = blockIdx.x * 256 + threadIdx.x;
  if (i < n) p[i] = 0.0f;
}

// ---------------------------------------------------------------------------
// Kernel 0b: pack weights to bf16, padded to 128x128, [n][k] layout.
//   mats: 0=W_L 1=W_hw 2..4=w_ih(r,z,n) 5..7=w_hh(r,z,n)
// ---------------------------------------------------------------------------
__global__ void k_pack(const float* __restrict__ W_L, const float* __restrict__ W_hw,
                       const float* __restrict__ w_ih, const float* __restrict__ w_hh,
                       __bf16* __restrict__ wpack) {
  const int idx = blockIdx.x * 256 + threadIdx.x;
  if (idx >= N_MATS * MAT_ELEMS) return;
  const int m = idx >> 14;
  const int n = (idx >> 7) & 127;
  const int k = idx & 127;
  float v = 0.0f;
  if (n < D_DIM && k < D_DIM) {
    switch (m) {
      case 0: v = W_L[n * D_DIM + k]; break;
      case 1: v = W_hw[n * D_DIM + k]; break;
      case 2: v = w_ih[(0   + n) * D_DIM + k]; break;
      case 3: v = w_ih[(120 + n) * D_DIM + k]; break;
      case 4: v = w_ih[(240 + n) * D_DIM + k]; break;
      case 5: v = w_hh[(0   + n) * D_DIM + k]; break;
      case 6: v = w_hh[(120 + n) * D_DIM + k]; break;
      case 7: v = w_hh[(240 + n) * D_DIM + k]; break;
    }
  }
  wpack[idx] = (__bf16)v;
}

// ---------------------------------------------------------------------------
// Kernel 1: va pass.  s = va @ W_L^T + b_L  (WMMA), plus per-graph atomics:
//   sum_s[g] += s,  sumSig[g] += sigmoid(s),  cnt[g] += 1.
// One wave handles a 16-node tile; 8 waves / block.
// ---------------------------------------------------------------------------
__global__ void __launch_bounds__(256) k_va(
    const float* __restrict__ va, const float* __restrict__ bL,
    const __bf16* __restrict__ wpack, const int* __restrict__ seg_a,
    float* __restrict__ s, float* __restrict__ sum_s,
    float* __restrict__ sumSig, float* __restrict__ cnt, int Na) {
  __shared__ __bf16 wl[MAT_ELEMS];
  lds_fill(wl, wpack, MAT_ELEMS * 2);   // matrix 0 = W_L
  __syncthreads();

  const int wave = threadIdx.x >> 5;
  const int lane = threadIdx.x & 31;
  const long tile = (long)blockIdx.x * 8 + wave;
  if (tile * 16 >= Na) return;
  const long node0 = tile * 16;

  const int col = lane & 15, half = lane >> 4;
  const float* rp = va + (node0 + col) * D_DIM;

  v16bf a[4];
#pragma unroll
  for (int t = 0; t < 4; ++t) a[t] = a_frag_f32(rp, 32 * t, D_DIM);

  if (lane < 16) atomicAdd(&cnt[seg_a[node0 + lane]], 1.0f);

  int gid[8];
#pragma unroll
  for (int v = 0; v < 8; ++v) gid[v] = seg_a[node0 + v + 8 * half];

#pragma unroll
  for (int nt = 0; nt < 8; ++nt) {
    v8f c = {};
#pragma unroll
    for (int t = 0; t < 4; ++t) c = wmma_bf16(a[t], b_frag(wl, 32 * t, nt), c);
    const int ch = nt * 16 + col;
    if (ch < D_DIM) {
      const float bias = bL[ch];
#pragma unroll
      for (int v = 0; v < 8; ++v) {
        const float val = c[v] + bias;
        const long node = node0 + v + 8 * half;
        s[node * D_DIM + ch] = val;
        atomicAdd(&sum_s[(long)gid[v] * D_DIM + ch], val);
        atomicAdd(&sumSig[(long)gid[v] * D_DIM + ch], sigf(val));
      }
    }
  }
}

// ---------------------------------------------------------------------------
// Kernel 2: SE block per graph: w = sigmoid(relu(avg @ Wse1^T) @ Wse2^T).
// ---------------------------------------------------------------------------
__global__ void __launch_bounds__(128) k_se(
    const float* __restrict__ sum_s, const float* __restrict__ cnt,
    const float* __restrict__ Wse1, const float* __restrict__ Wse2,
    float* __restrict__ w_se) {
  __shared__ float avg[D_DIM];
  __shared__ float r[R_DIM];
  const int g = blockIdx.x, t = threadIdx.x;
  const float c = fmaxf(cnt[g], 1.0f);
  if (t < D_DIM) avg[t] = sum_s[(long)g * D_DIM + t] * fast_rcp(c);
  __syncthreads();
  if (t < R_DIM) {
    float acc = 0.0f;
    for (int k = 0; k < D_DIM; ++k) acc += Wse1[t * D_DIM + k] * avg[k];
    r[t] = fmaxf(acc, 0.0f);
  }
  __syncthreads();
  if (t < D_DIM) {
    float acc = 0.0f;
#pragma unroll
    for (int j = 0; j < R_DIM; ++j) acc += Wse2[t * R_DIM + j] * r[j];
    w_se[(long)g * D_DIM + t] = sigf(acc);
  }
}

// ---------------------------------------------------------------------------
// Kernel 3: g_feat[g] += s_i * (s_i . w[g]) / 120   (one wave per node).
// ---------------------------------------------------------------------------
__global__ void __launch_bounds__(256) k_gfeat(
    const float* __restrict__ s, const float* __restrict__ w_se,
    const int* __restrict__ seg_a, float* __restrict__ g_feat, int Na) {
  const int lane = threadIdx.x & 31;
  const int wid  = (blockIdx.x * 256 + threadIdx.x) >> 5;
  const int nw   = (gridDim.x * 256) >> 5;
  for (long node = wid; node < Na; node += nw) {
    const int g = seg_a[node];
    const float* sp = s + node * D_DIM;
    const float* wp = w_se + (long)g * D_DIM;
    __builtin_prefetch(s + (node + nw) * D_DIM, 0, 0);
    float part = 0.0f;
    for (int c = lane; c < D_DIM; c += 32) part += sp[c] * wp[c];
#pragma unroll
    for (int off = 16; off > 0; off >>= 1) part += __shfl_xor(part, off, 32);
    const float m = part * (1.0f / (float)D_DIM);
    for (int c = lane; c < D_DIM; c += 32)
      atomicAdd(&g_feat[(long)g * D_DIM + c], sp[c] * m);
  }
}

// ---------------------------------------------------------------------------
// Kernel 4: pooled = sumSig * g_feat   (elementwise over G*120).
// ---------------------------------------------------------------------------
__global__ void k_pool(const float* __restrict__ sumSig,
                       const float* __restrict__ g_feat,
                       float* __restrict__ pooled, int n) {
  int i = blockIdx.x * 256 + threadIdx.x;
  if (i < n) pooled[i] = sumSig[i] * g_feat[i];
}

// ---------------------------------------------------------------------------
// Kernel 5: fused vb branch.  All 8 weight mats (256 KB bf16) + per-wave
// transpose stages (32 KB) in dynamic LDS.  288 WMMAs per 16-node tile.
// ---------------------------------------------------------------------------
__global__ void __launch_bounds__(256) k_vb(
    const float* __restrict__ vb, const __bf16* __restrict__ wpack,
    const float* __restrict__ bL, const float* __restrict__ bhw,
    const float* __restrict__ bih, const float* __restrict__ bhh,
    const float* __restrict__ pooled, const int* __restrict__ seg_b,
    float* __restrict__ out, int Nb) {
  extern __shared__ __bf16 lds[];
  lds_fill(lds, wpack, N_MATS * MAT_ELEMS * 2);
  __syncthreads();

  const __bf16* Wl  = lds;
  const __bf16* Whw = lds + 1 * MAT_ELEMS;
  const __bf16* Wr  = lds + 2 * MAT_ELEMS;
  const __bf16* Wz  = lds + 3 * MAT_ELEMS;
  const __bf16* Wn  = lds + 4 * MAT_ELEMS;
  const __bf16* Whr = lds + 5 * MAT_ELEMS;
  const __bf16* Whz = lds + 6 * MAT_ELEMS;
  const __bf16* Whn = lds + 7 * MAT_ELEMS;

  const int wave = threadIdx.x >> 5;
  const int lane = threadIdx.x & 31;
  __bf16* stage = lds + N_MATS * MAT_ELEMS + wave * (16 * D_PAD);

  const long tile = (long)blockIdx.x * 8 + wave;
  if (tile * 16 >= Nb) return;
  const long node0 = tile * 16;
  const int col = lane & 15, half = lane >> 4;

  // ---- GEMM 1: vb_l = vb @ W_L^T + b_L
  const float* rp = vb + (node0 + col) * D_DIM;
  v16bf a_vb[4];
#pragma unroll
  for (int t = 0; t < 4; ++t) a_vb[t] = a_frag_f32(rp, 32 * t, D_DIM);

  v8f vbl[8];
#pragma unroll
  for (int nt = 0; nt < 8; ++nt) {
    v8f c = {};
#pragma unroll
    for (int t = 0; t < 4; ++t) c = wmma_bf16(a_vb[t], b_frag(Wl, 32 * t, nt), c);
    const int ch = nt * 16 + col;
    const float bias = (ch < D_DIM) ? bL[ch] : 0.0f;
#pragma unroll
    for (int v = 0; v < 8; ++v) c[v] += bias;
    vbl[nt] = c;
  }

  // transpose vb_l into A layout via wave-private LDS stage
  c_to_stage(stage, vbl);
  asm volatile("s_wait_dscnt 0" ::: "memory");
  v16bf a_vbl[4];
#pragma unroll
  for (int t = 0; t < 4; ++t) a_vbl[t] = a_frag_lds(stage, 32 * t);

  // vvb = tanh(vb_l) directly in A layout
  v16bf a_vvb[4];
#pragma unroll
  for (int t = 0; t < 4; ++t) {
#pragma unroll
    for (int e = 0; e < 16; ++e)
      a_vvb[t][e] = (__bf16)tanh_fast((float)a_vbl[t][e]);
  }

  // va_b = pooled[seg_b[node]] as A fragments
  const float* prp = pooled + (long)seg_b[node0 + col] * D_DIM;
  v16bf a_vab[4];
#pragma unroll
  for (int t = 0; t < 4; ++t) a_vab[t] = a_frag_f32(prp, 32 * t, D_DIM);

  // ---- Highway: a = va_b@Whw^T+b, b = vvb@Whw^T+b, h = z*b + (1-z)*a
  v8f hc[8];
#pragma unroll
  for (int nt = 0; nt < 8; ++nt) {
    v8f ac = {}, bc = {};
#pragma unroll
    for (int t = 0; t < 4; ++t) ac = wmma_bf16(a_vab[t], b_frag(Whw, 32 * t, nt), ac);
#pragma unroll
    for (int t = 0; t < 4; ++t) bc = wmma_bf16(a_vvb[t], b_frag(Whw, 32 * t, nt), bc);
    const int ch = nt * 16 + col;
    const float bias = (ch < D_DIM) ? bhw[ch] : 0.0f;
    v8f h;
#pragma unroll
    for (int v = 0; v < 8; ++v) {
      const float av = ac[v] + bias, bv = bc[v] + bias;
      const float z = sigf(av + bv);
      h[v] = z * bv + (1.0f - z) * av;
    }
    hc[nt] = h;
  }

  // transpose h into A layout (reuse stage)
  c_to_stage(stage, hc);
  asm volatile("s_wait_dscnt 0" ::: "memory");
  v16bf a_h[4];
#pragma unroll
  for (int t = 0; t < 4; ++t) a_h[t] = a_frag_lds(stage, 32 * t);

  // ---- GRU: r,z,n gates (6 GEMMs per channel tile), then output.
#pragma unroll
  for (int ct = 0; ct < 8; ++ct) {
    v8f gir = {}, giz = {}, gin = {}, ghr = {}, ghz = {}, ghn = {};
#pragma unroll
    for (int t = 0; t < 4; ++t) {
      gir = wmma_bf16(a_vbl[t], b_frag(Wr,  32 * t, ct), gir);
      giz = wmma_bf16(a_vbl[t], b_frag(Wz,  32 * t, ct), giz);
      gin = wmma_bf16(a_vbl[t], b_frag(Wn,  32 * t, ct), gin);
      ghr = wmma_bf16(a_h[t],   b_frag(Whr, 32 * t, ct), ghr);
      ghz = wmma_bf16(a_h[t],   b_frag(Whz, 32 * t, ct), ghz);
      ghn = wmma_bf16(a_h[t],   b_frag(Whn, 32 * t, ct), ghn);
    }
    const int ch = ct * 16 + col;
    if (ch < D_DIM) {
      const float bir = bih[ch],       bhr = bhh[ch];
      const float biz = bih[120 + ch], bhz = bhh[120 + ch];
      const float bin = bih[240 + ch], bhn = bhh[240 + ch];
#pragma unroll
      for (int v = 0; v < 8; ++v) {
        const float r_ = sigf(gir[v] + bir + ghr[v] + bhr);
        const float zz = sigf(giz[v] + biz + ghz[v] + bhz);
        const float nn = tanh_fast(gin[v] + bin + r_ * (ghn[v] + bhn));
        const long node = node0 + v + 8 * half;
        out[node * D_DIM + ch] = (1.0f - zz) * nn + zz * hc[ct][v];
      }
    }
  }
}

// ---------------------------------------------------------------------------
// Host launcher
// ---------------------------------------------------------------------------
extern "C" void kernel_launch(void* const* d_in, const int* in_sizes, int n_in,
                              void* d_out, int out_size, void* d_ws, size_t ws_size,
                              hipStream_t stream) {
  const float* va    = (const float*)d_in[0];
  const float* vb    = (const float*)d_in[1];
  const float* W_L   = (const float*)d_in[2];
  const float* b_L   = (const float*)d_in[3];
  const float* W_se1 = (const float*)d_in[4];
  const float* W_se2 = (const float*)d_in[5];
  const float* W_hw  = (const float*)d_in[6];
  const float* b_hw  = (const float*)d_in[7];
  const float* w_ih  = (const float*)d_in[8];
  const float* w_hh  = (const float*)d_in[9];
  const float* b_ih  = (const float*)d_in[10];
  const float* b_hh  = (const float*)d_in[11];
  const int*   seg_a = (const int*)d_in[12];
  const int*   seg_b = (const int*)d_in[13];

  const int Na = in_sizes[12];
  const int Nb = in_sizes[13];
  const int G  = G_MAX;   // num_graphs fixed by setup

  // workspace layout (floats): s | sum_s | sumSig | g_feat | cnt | pooled | w_se | wpack(bf16)
  float* ws      = (float*)d_ws;
  float* s       = ws;
  float* sum_s   = s      + (size_t)Na * D_DIM;
  float* sumSig  = sum_s  + (size_t)G * D_DIM;
  float* g_feat  = sumSig + (size_t)G * D_DIM;
  float* cntp    = g_feat + (size_t)G * D_DIM;
  float* pooled  = cntp   + (size_t)G;
  float* w_se    = pooled + (size_t)G * D_DIM;
  __bf16* wpack  = (__bf16*)(w_se + (size_t)G * D_DIM);

  const int nzero = 3 * G * D_DIM + G;   // sum_s, sumSig, g_feat, cnt contiguous
  k_zero<<<(nzero + 255) / 256, 256, 0, stream>>>(sum_s, nzero);

  const int npack = N_MATS * MAT_ELEMS;
  k_pack<<<(npack + 255) / 256, 256, 0, stream>>>(W_L, W_hw, w_ih, w_hh, wpack);

  const int tilesA = (Na + 15) / 16;
  k_va<<<(tilesA + 7) / 8, 256, 0, stream>>>(va, b_L, wpack, seg_a,
                                             s, sum_s, sumSig, cntp, Na);

  k_se<<<G, 128, 0, stream>>>(sum_s, cntp, W_se1, W_se2, w_se);

  k_gfeat<<<2048, 256, 0, stream>>>(s, w_se, seg_a, g_feat, Na);

  k_pool<<<(G * D_DIM + 255) / 256, 256, 0, stream>>>(sumSig, g_feat, pooled, G * D_DIM);

  const int tilesB = (Nb + 15) / 16;
  const size_t lds_bytes = (size_t)(N_MATS * MAT_ELEMS + 8 * 16 * D_PAD) * sizeof(__bf16);
  k_vb<<<(tilesB + 7) / 8, 256, lds_bytes, stream>>>(vb, wpack, b_L, b_hw, b_ih, b_hh,
                                                     pooled, seg_b, (float*)d_out, Nb);
}
